// BiasedMultiHeadAttention_20074677141745
// MI455X (gfx1250) — matrix operand order repaired and checked by reading, not compiled, not run on hardware
//
#include <hip/hip_runtime.h>

// ---------------------------------------------------------------------------
// Types for CDNA5 WMMA (wave32, 16x16x32 bf16 -> f32)
// ---------------------------------------------------------------------------
typedef __attribute__((ext_vector_type(16))) __bf16       v16bf;
typedef __attribute__((ext_vector_type(8)))  float        v8f;
typedef __attribute__((ext_vector_type(4)))  unsigned int v4u;
typedef int v4i_vs __attribute__((vector_size(16)));       // matches builtin param

union Frag16 { v16bf v; v4u u[2]; };

#define SEQ 4096
#define DMODEL 1024
#define NHEAD 16
#define NEGV -1000000000.0f

__device__ __forceinline__ unsigned short f32_to_bf16(float f) {
  union { float f; unsigned int u; } c; c.f = f;
  unsigned int x = c.u;
  x += 0x7fffu + ((x >> 16) & 1u);   // round-to-nearest-even
  return (unsigned short)(x >> 16);
}

__device__ __forceinline__ unsigned int pack_bf16x2(float a, float b) {
  return (unsigned int)f32_to_bf16(a) | ((unsigned int)f32_to_bf16(b) << 16);
}

// ---------------------------------------------------------------------------
// Async global -> LDS copy (CDNA5 path, ASYNCcnt-tracked); sync fallback.
// Builtin signature (from hipcc diagnostic): param0 = v4i addrspace(1)*,
// i.e. (global_src, lds_dst, imm_offset, imm_cpol).
// ---------------------------------------------------------------------------
#if __has_builtin(__builtin_amdgcn_global_load_async_to_lds_b128)
#define HAVE_ASYNC_LDS 1
#else
#define HAVE_ASYNC_LDS 0
#endif

__device__ __forceinline__ void async_copy_b128(const unsigned short* g,
                                                unsigned short* l) {
#if HAVE_ASYNC_LDS
  __builtin_amdgcn_global_load_async_to_lds_b128(
      (__attribute__((address_space(1))) v4i_vs*)g,
      (__attribute__((address_space(3))) v4i_vs*)l, 0, 0);
#else
  *(v4u*)l = *(const v4u*)g;
#endif
}

__device__ __forceinline__ void async_join() {
#if HAVE_ASYNC_LDS
#if __has_builtin(__builtin_amdgcn_s_wait_asynccnt)
  __builtin_amdgcn_s_wait_asynccnt(0);
#else
  asm volatile("s_wait_asynccnt 0" ::: "memory");
#endif
#endif
}

// ---------------------------------------------------------------------------
// Kernel 1: f32 -> bf16 conversion, 4 elems/thread (float4 in, uint2 out)
// ---------------------------------------------------------------------------
__global__ __launch_bounds__(256)
void convert_bf16_kernel(const float* __restrict__ src,
                         unsigned short* __restrict__ dst, int n4) {
  int i = blockIdx.x * blockDim.x + threadIdx.x;
  const int stride = gridDim.x * blockDim.x;
  const float4* s4 = (const float4*)src;
  uint2* d2 = (uint2*)dst;
  for (; i < n4; i += stride) {
    float4 f = s4[i];
    uint2 o;
    o.x = pack_bf16x2(f.x, f.y);
    o.y = pack_bf16x2(f.z, f.w);
    d2[i] = o;
  }
}

// ---------------------------------------------------------------------------
// Kernel 2: C = A(MxK) * B(NxK)^T, bf16 in, f32 accumulate via WMMA.
// Block: 256 thr = 8 waves. Block tile 256(M) x 64(N); wave tile 32 x 64
// (2 A-frags x 4 B-frags -> 8 WMMAs per K-step; B reused across both M tiles).
// MODE 0: bf16 row-major  | MODE 1: bf16 transposed (V^T) | MODE 2: f32 row-major
// ---------------------------------------------------------------------------
template <int MODE>
__global__ __launch_bounds__(256)
void gemm_bt_kernel(const unsigned short* __restrict__ A,
                    const unsigned short* __restrict__ B,
                    void* __restrict__ Cout,
                    int M, int N, int K) {
  const int lane = threadIdx.x & 31;
  const int wave = threadIdx.x >> 5;
  const int g  = lane >> 4;
  const int ln = lane & 15;
  const int m0 = blockIdx.x * 256 + wave * 32;
  const int n0 = blockIdx.y * 64;

  v8f acc[2][4] = {{{}, {}, {}, {}}, {{}, {}, {}, {}}};

  const unsigned short* Arow0 = A + (size_t)(m0 + ln) * K;
  const unsigned short* Arow1 = A + (size_t)(m0 + 16 + ln) * K;
  for (int kk = 0; kk < K; kk += 32) {
    Frag16 a0, a1;
    a0.u[0] = *(const v4u*)(Arow0 + kk + 8 * g);
    a0.u[1] = *(const v4u*)(Arow0 + kk + 16 + 8 * g);
    a1.u[0] = *(const v4u*)(Arow1 + kk + 8 * g);
    a1.u[1] = *(const v4u*)(Arow1 + kk + 16 + 8 * g);
#pragma unroll
    for (int t = 0; t < 4; ++t) {
      Frag16 b;
      const unsigned short* Brow =
          B + (size_t)(n0 + t * 16 + ln) * K + kk + 8 * g;
      b.u[0] = *(const v4u*)(Brow);
      b.u[1] = *(const v4u*)(Brow + 16);
      acc[0][t] = __builtin_amdgcn_wmma_f32_16x16x32_bf16(
          false, a0.v, false, b.v, (short)0, acc[0][t], false, false);
      acc[1][t] = __builtin_amdgcn_wmma_f32_16x16x32_bf16(
          false, a1.v, false, b.v, (short)0, acc[1][t], false, false);
    }
  }

#pragma unroll
  for (int mt = 0; mt < 2; ++mt) {
    const int mb = m0 + mt * 16;
    if (MODE == 0) {
      unsigned short* C = (unsigned short*)Cout;
#pragma unroll
      for (int t = 0; t < 4; ++t) {
        const int nc = n0 + t * 16 + ln;
#pragma unroll
        for (int r = 0; r < 8; ++r)
          C[(size_t)(mb + 8 * g + r) * N + nc] = f32_to_bf16(acc[mt][t][r]);
      }
    } else if (MODE == 1) {
      unsigned short* Ct = (unsigned short*)Cout;
#pragma unroll
      for (int t = 0; t < 4; ++t) {
        const int nc = n0 + t * 16 + ln;
        v4u pv;
#pragma unroll
        for (int r = 0; r < 4; ++r)
          pv[r] = pack_bf16x2(acc[mt][t][2 * r], acc[mt][t][2 * r + 1]);
        *(v4u*)(Ct + (size_t)nc * M + mb + 8 * g) = pv;  // 16B column store
      }
    } else {
      float* C = (float*)Cout;
#pragma unroll
      for (int t = 0; t < 4; ++t) {
        const int nc = n0 + t * 16 + ln;
#pragma unroll
        for (int r = 0; r < 8; ++r)
          C[(size_t)(mb + 8 * g + r) * N + nc] = acc[mt][t][r];
      }
    }
  }
}

// ---------------------------------------------------------------------------
// Kernel 3: flash attention. Block = 128 thr = 4 waves, one head, 64 query
// rows (16 per wave). Per 32-key step the block cooperatively stages the
// K tile (32x64 bf16) and V^T tile (64x32 bf16) into LDS with async
// global->LDS loads (ASYNCcnt), then every wave builds its WMMA fragments
// from LDS. All waves run a uniform step count; causal masking neutralizes
// tail steps for low-q waves, keeping __syncthreads() safe.
// ---------------------------------------------------------------------------
__global__ __launch_bounds__(128)
void attn_kernel(const unsigned short* __restrict__ Q,   // S x 1024 bf16
                 const unsigned short* __restrict__ Kb,  // S x 1024 bf16
                 const unsigned short* __restrict__ Vt,  // 1024 x S bf16 (V^T)
                 const float* __restrict__ bias,         // S x S f32
                 const unsigned char* __restrict__ mask, // S bool
                 unsigned short* __restrict__ O) {       // S x 1024 bf16
  __shared__ unsigned short sK[32 * 64];   // [key in tile][dk]
  __shared__ unsigned short sV[64 * 32];   // [dv][key in tile]
  __shared__ unsigned short sP[4 * 512];   // per-wave 16x32 P transpose

  const int tid  = threadIdx.x;
  const int lane = tid & 31;
  const int wave = tid >> 5;
  const int g  = lane >> 4;
  const int ln = lane & 15;
  const int head = blockIdx.y;
  const int qb = blockIdx.x * 64;
  const int q0 = qb + wave * 16;

  // Q A-fragments (two halves of DK=64)
  Frag16 qa[2];
  const unsigned short* Qh = Q + (size_t)(q0 + ln) * DMODEL + head * 64;
#pragma unroll
  for (int h2 = 0; h2 < 2; ++h2) {
    qa[h2].u[0] = *(const v4u*)(Qh + h2 * 32 + 8 * g);
    qa[h2].u[1] = *(const v4u*)(Qh + h2 * 32 + 16 + 8 * g);
  }

  v8f o[4] = {{}, {}, {}, {}};
  float st_m[8], st_l[8];
#pragma unroll
  for (int r = 0; r < 8; ++r) { st_m[r] = -3.0e38f; st_l[r] = 0.f; }

  unsigned short* myP = sP + wave * 512;
  const int nsteps = (qb + 95) / 32;  // covers keys 0..qb+63; never exceeds S

  for (int step = 0; step < nsteps; ++step) {
    const int j0 = step * 32;

    // ---- cooperative async staging of K and V^T tiles into LDS ----
    {
      const int kr = tid >> 2;                 // 0..31 key row
      const int kc = (tid & 3) * 16;           // element chunk
      const unsigned short* gp =
          Kb + (size_t)(j0 + kr) * DMODEL + head * 64 + kc;
      unsigned short* lp = sK + kr * 64 + kc;
      async_copy_b128(gp, lp);
      async_copy_b128(gp + 8, lp + 8);

      const int vd = tid >> 1;                 // 0..63 dv row
      const int vc = (tid & 1) * 16;
      const unsigned short* gv =
          Vt + (size_t)(head * 64 + vd) * SEQ + j0 + vc;
      unsigned short* lv = sV + vd * 32 + vc;
      async_copy_b128(gv, lv);
      async_copy_b128(gv + 8, lv + 8);
    }
    async_join();
    __syncthreads();

    // ---- scores: two 16-key tiles from LDS K ----
    v8f sc[2];
#pragma unroll
    for (int t = 0; t < 2; ++t) {
      v8f c = {};
      const unsigned short* Krow = sK + (t * 16 + ln) * 64;
#pragma unroll
      for (int h2 = 0; h2 < 2; ++h2) {
        Frag16 kf;
        const unsigned short* p = Krow + h2 * 32 + 8 * g;
        kf.u[0] = *(const v4u*)(p);
        kf.u[1] = *(const v4u*)(p + 16);
        c = __builtin_amdgcn_wmma_f32_16x16x32_bf16(
            false, qa[h2].v, false, kf.v, (short)0, c, false, false);
      }
      sc[t] = c;
    }

    // ---- bias + key-mask + causal (reference: mask replaces, causal adds) ----
#pragma unroll
    for (int t = 0; t < 2; ++t) {
      const int j = j0 + t * 16 + ln;
      const int mskd = mask[j];
      const float* bcol = bias + j;
#pragma unroll
      for (int r = 0; r < 8; ++r) {
        const int q = q0 + 8 * g + r;
        float s = sc[t][r] * 0.125f + bcol[(size_t)q * SEQ];
        if (mskd) s = NEGV;
        if (j > q) s += NEGV;
        sc[t][r] = s;
      }
    }

    // ---- online softmax (rows span 16 lanes of a half-wave) ----
    float p0v[8], p1v[8], alpha[8];
#pragma unroll
    for (int r = 0; r < 8; ++r) {
      float v = fmaxf(sc[0][r], sc[1][r]);
      v = fmaxf(v, __shfl_xor(v, 1, 32));
      v = fmaxf(v, __shfl_xor(v, 2, 32));
      v = fmaxf(v, __shfl_xor(v, 4, 32));
      v = fmaxf(v, __shfl_xor(v, 8, 32));
      const float nm = fmaxf(st_m[r], v);
      alpha[r] = __expf(st_m[r] - nm);
      st_m[r] = nm;
      const float p0 = __expf(sc[0][r] - nm);
      const float p1 = __expf(sc[1][r] - nm);
      p0v[r] = p0; p1v[r] = p1;
      float rs = p0 + p1;
      rs += __shfl_xor(rs, 1, 32);
      rs += __shfl_xor(rs, 2, 32);
      rs += __shfl_xor(rs, 4, 32);
      rs += __shfl_xor(rs, 8, 32);
      st_l[r] = st_l[r] * alpha[r] + rs;
    }

#pragma unroll
    for (int f = 0; f < 4; ++f)
#pragma unroll
      for (int r = 0; r < 8; ++r) o[f][r] *= alpha[r];

    // ---- transpose P (C-layout) -> A-fragment layout via per-wave LDS ----
#pragma unroll
    for (int r = 0; r < 8; ++r) {
      myP[(8 * g + r) * 32 + ln]      = f32_to_bf16(p0v[r]);
      myP[(8 * g + r) * 32 + 16 + ln] = f32_to_bf16(p1v[r]);
    }
    asm volatile("" ::: "memory");  // keep ds_store before ds_load (HW in-order per wave)
    Frag16 pa;
    {
      const unsigned short* p = myP + ln * 32 + 8 * g;
      pa.u[0] = *(const v4u*)(p);
      pa.u[1] = *(const v4u*)(p + 16);
    }

    // ---- O += P(16x32) @ V(32x64) from LDS V^T tile ----
#pragma unroll
    for (int f = 0; f < 4; ++f) {
      Frag16 vb;
      const unsigned short* p = sV + (f * 16 + ln) * 32 + 8 * g;
      vb.u[0] = *(const v4u*)(p);
      vb.u[1] = *(const v4u*)(p + 16);
      o[f] = __builtin_amdgcn_wmma_f32_16x16x32_bf16(
          false, pa.v, false, vb.v, (short)0, o[f], false, false);
    }

    __syncthreads();  // all waves done reading sK/sV before next staging
  }

  // ---- normalize and store attention output (bf16, S x H*DV layout) ----
#pragma unroll
  for (int r = 0; r < 8; ++r) st_l[r] = 1.0f / st_l[r];
#pragma unroll
  for (int f = 0; f < 4; ++f)
#pragma unroll
    for (int r = 0; r < 8; ++r)
      O[(size_t)(q0 + 8 * g + r) * DMODEL + head * 64 + f * 16 + ln] =
          f32_to_bf16(o[f][r] * st_l[r]);
}

// ---------------------------------------------------------------------------
// Host-side orchestration
// ---------------------------------------------------------------------------
extern "C" void kernel_launch(void* const* d_in, const int* in_sizes, int n_in,
                              void* d_out, int out_size, void* d_ws, size_t ws_size,
                              hipStream_t stream) {
  const float*         x    = (const float*)d_in[0];
  const float*         bias = (const float*)d_in[1];
  const unsigned char* mask = (const unsigned char*)d_in[2];  // jax bool: 1B/elem
  const float*         Wq   = (const float*)d_in[3];
  const float*         Wk   = (const float*)d_in[4];
  const float*         Wv   = (const float*)d_in[5];
  const float*         Wo   = (const float*)d_in[6];
  float* out = (float*)d_out;

  const size_t SD = (size_t)SEQ * DMODEL;     // 4 Mi elems
  const size_t WW = (size_t)DMODEL * DMODEL;  // 1 Mi elems

  unsigned short* ws = (unsigned short*)d_ws;
  unsigned short* xb   = ws;            // bf16 x                 [SD]
  unsigned short* wqb  = xb  + SD;      // bf16 Wq                [WW]
  unsigned short* wkb  = wqb + WW;
  unsigned short* wvb  = wkb + WW;
  unsigned short* wob  = wvb + WW;
  unsigned short* Qb   = wob + WW;      // Q = x Wq^T  (S x 1024) [SD]
  unsigned short* Kbuf = Qb  + SD;      // K = x Wk^T  (S x 1024) [SD]
  unsigned short* Vtb  = Kbuf+ SD;      // V^T         (1024 x S) [SD]
  unsigned short* Ob   = Vtb + SD;      // attn out    (S x 1024) [SD]
  const size_t need = (5 * SD + 4 * WW) * sizeof(unsigned short);
  if (ws_size < need) return;

  // 1) f32 -> bf16 (4 elems / thread)
  convert_bf16_kernel<<<1024, 256, 0, stream>>>(x,  xb,  (int)(SD / 4));
  convert_bf16_kernel<<<256,  256, 0, stream>>>(Wq, wqb, (int)(WW / 4));
  convert_bf16_kernel<<<256,  256, 0, stream>>>(Wk, wkb, (int)(WW / 4));
  convert_bf16_kernel<<<256,  256, 0, stream>>>(Wv, wvb, (int)(WW / 4));
  convert_bf16_kernel<<<256,  256, 0, stream>>>(Wo, wob, (int)(WW / 4));

  // 2) projections: Q, K row-major; V stored transposed
  dim3 ggrid(SEQ / 256, DMODEL / 64);
  gemm_bt_kernel<0><<<ggrid, 256, 0, stream>>>(xb, wqb, Qb,   SEQ, DMODEL, DMODEL);
  gemm_bt_kernel<0><<<ggrid, 256, 0, stream>>>(xb, wkb, Kbuf, SEQ, DMODEL, DMODEL);
  gemm_bt_kernel<1><<<ggrid, 256, 0, stream>>>(xb, wvb, Vtb,  SEQ, DMODEL, DMODEL);

  // 3) flash attention (grid: 64 query blocks x 16 heads, 4 waves/block)
  attn_kernel<<<dim3(SEQ / 64, NHEAD), 128, 0, stream>>>(Qb, Kbuf, Vtb, bias, mask, Ob);

  // 4) output projection: out = O Wo^T (f32 store)
  gemm_bt_kernel<2><<<ggrid, 256, 0, stream>>>(Ob, wob, out, SEQ, DMODEL, DMODEL);
}
